// CharacterAwareAttention_7172595384972
// MI455X (gfx1250) — matrix operand order repaired and checked by reading, not compiled, not run on hardware
//
#include <hip/hip_runtime.h>
#include <hip/hip_bf16.h>

typedef __attribute__((ext_vector_type(16))) _Float16 v16h;
typedef __attribute__((ext_vector_type(8)))  _Float16 v8h;
typedef __attribute__((ext_vector_type(8)))  float    v8f;

#define LN_EPS 1e-5f

__device__ __forceinline__ v8f wmma_f16(v16h a, v16h b, v8f c) {
  return __builtin_amdgcn_wmma_f32_16x16x32_f16(false, a, false, b, (short)0, c, false, false);
}

__device__ __forceinline__ v8f zero8() {
  v8f z = {0.f,0.f,0.f,0.f,0.f,0.f,0.f,0.f};
  return z;
}

// A-operand (16x32 f16): lane-half 0 holds K {0..7, 16..23}; lane-half 1 holds
// K {8..15, 24..31}. Caller passes p = rowbase + kd + 8*half.
__device__ __forceinline__ v16h ldA16(const _Float16* p) {
  v8h a = *(const v8h*)(p);
  v8h b = *(const v8h*)(p + 16);
  v16h r;
#pragma unroll
  for (int i = 0; i < 8; ++i) { r[i] = a[i]; r[i + 8] = b[i]; }
  return r;
}

// ---------------------------------------------------------------------------
// 0) elementwise f32 -> f16 conversion (8 elements/thread, 32 B in / 16 B out)
// ---------------------------------------------------------------------------
__global__ void cvt_kernel(const float* __restrict__ in, _Float16* __restrict__ out, size_t n) {
  size_t i = ((size_t)blockIdx.x * blockDim.x + threadIdx.x) * 8;
  if (i >= n) return;
  v8f v = *(const v8f*)(in + i);
  v8h o;
#pragma unroll
  for (int j = 0; j < 8; ++j) o[j] = (_Float16)v[j];
  *(v8h*)(out + i) = o;
}

// ---------------------------------------------------------------------------
// 1) LayerNorm of queries (f32 in -> f16 out)
// ---------------------------------------------------------------------------
__global__ void ln_kernel(const float* __restrict__ q, const float* __restrict__ gamma,
                          const float* __restrict__ beta, _Float16* __restrict__ x, int Dn) {
  const int row = blockIdx.x;
  const float* p = q + (size_t)row * Dn;
  float s = 0.f, ss = 0.f;
  for (int i = threadIdx.x; i < Dn; i += blockDim.x) { float v = p[i]; s += v; ss += v * v; }
#pragma unroll
  for (int off = 16; off > 0; off >>= 1) { s += __shfl_xor(s, off); ss += __shfl_xor(ss, off); }
  __shared__ float sh[16], sh2[16];
  const int wid = threadIdx.x >> 5, lid = threadIdx.x & 31, nw = blockDim.x >> 5;
  if (lid == 0) { sh[wid] = s; sh2[wid] = ss; }
  __syncthreads();
  if (wid == 0) {
    s = (lid < nw) ? sh[lid] : 0.f;
    ss = (lid < nw) ? sh2[lid] : 0.f;
#pragma unroll
    for (int off = 16; off > 0; off >>= 1) { s += __shfl_xor(s, off); ss += __shfl_xor(ss, off); }
    if (lid == 0) { sh[0] = s; sh2[0] = ss; }
  }
  __syncthreads();
  const float mu = sh[0] / (float)Dn;
  const float var = sh2[0] / (float)Dn - mu * mu;
  const float inv = rsqrtf(var + LN_EPS);
  _Float16* o = x + (size_t)row * Dn;
  for (int i = threadIdx.x; i < Dn; i += blockDim.x)
    o[i] = (_Float16)((p[i] - mu) * inv * gamma[i] + beta[i]);
}

// ---------------------------------------------------------------------------
// 2) Per-query mask intervals: union of [c0,c1) U [p0,p1) U [n0,n1) -> +1.0
// ---------------------------------------------------------------------------
__global__ void intervals_kernel(const int* __restrict__ wb, const int* __restrict__ cb,
                                 int* __restrict__ iv, int Qn, int nW) {
  const int q = blockIdx.x * blockDim.x + threadIdx.x;
  if (q >= Qn) return;
  int lo = 0, hi = nW + 1;
  while (lo < hi) { int mid = (lo + hi) >> 1; if (wb[mid] <= q) lo = mid + 1; else hi = mid; }
  int i = lo - 1;
  if (i < 0) i = 0;
  if (i > nW - 1) i = nW - 1;
  const int ws = wb[i], we = wb[i + 1];
  const bool valid = (q >= ws) && (q < we);
  int c0 = 0, c1 = 0, p0 = 0, p1 = 0, n0 = 0, n1 = 0;
  if (valid) {
    c0 = cb[ws]; c1 = cb[we - 1];
    if (i > 0) { p0 = wb[i - 1]; p1 = ws; }
    if (i < nW - 1) { n0 = we; int ip2 = i + 2; if (ip2 > nW) ip2 = nW; n1 = wb[ip2]; }
  }
  iv[q * 6 + 0] = c0; iv[q * 6 + 1] = c1;
  iv[q * 6 + 2] = p0; iv[q * 6 + 3] = p1;
  iv[q * 6 + 4] = n0; iv[q * 6 + 5] = n1;
}

__device__ __forceinline__ float maskv(int k, int c0, int c1, int p0, int p1, int n0, int n1) {
  return ((k >= c0 && k < c1) || (k >= p0 && k < p1) || (k >= n0 && k < n1)) ? 1.0f : 0.0f;
}

// ---------------------------------------------------------------------------
// 3) f16 WMMA GEMM: C[M,N] = alpha*(A[M,K]·W[N,K]^T + bias[N]) (+resid)
//    16x64 output strip per wave (A-tile reused across 4 WMMAs per k-step).
//    mode 0: f16 row-major. mode 1: f16 V^T layout o16[(b*N+n)*vtK + k].
//    mode 2: f32 o32 = resid + value.
// ---------------------------------------------------------------------------
__global__ void __launch_bounds__(128)
gemm_f16_wmma(const _Float16* __restrict__ A, const _Float16* __restrict__ W,
              const float* __restrict__ bias, float alpha,
              int M, int N, int Kd,
              _Float16* __restrict__ o16, float* __restrict__ o32,
              const float* __restrict__ resid, int mode, int vtK) {
  const int lane = threadIdx.x & 31;
  const int wave = threadIdx.x >> 5;
  const int t = blockIdx.x * (blockDim.x >> 5) + wave;
  const int ngrp = N >> 6;            // groups of 4 consecutive 16-wide N tiles
  const int ntm = M >> 4;
  if (t >= ntm * ngrp) return;        // uniform per wave: EXEC stays all-ones
  const int tm = t / ngrp, tg = t - tm * ngrp;
  const int half = lane >> 4;
  const int lq = lane & 15;
  const int row = tm * 16 + lq;
  const int nbase = tg * 64;
  const _Float16* arow = A + (size_t)row * Kd + half * 8;
  // B-operand rows: lanes 0-15 hold K {0..7, 16..23}, lanes 16-31 hold K {8..15, 24..31}
  const _Float16* w0 = W + (size_t)(nbase + 0  + lq) * Kd + half * 8;
  const _Float16* w1 = W + (size_t)(nbase + 16 + lq) * Kd + half * 8;
  const _Float16* w2 = W + (size_t)(nbase + 32 + lq) * Kd + half * 8;
  const _Float16* w3 = W + (size_t)(nbase + 48 + lq) * Kd + half * 8;
  v8f accs[4] = {zero8(), zero8(), zero8(), zero8()};
  for (int kd = 0; kd < Kd; kd += 32) {
    const v16h a = ldA16(arow + kd);
    accs[0] = wmma_f16(a, ldA16(w0 + kd), accs[0]);
    accs[1] = wmma_f16(a, ldA16(w1 + kd), accs[1]);
    accs[2] = wmma_f16(a, ldA16(w2 + kd), accs[2]);
    accs[3] = wmma_f16(a, ldA16(w3 + kd), accs[3]);
  }
#pragma unroll
  for (int j = 0; j < 4; ++j) {
    const int n = nbase + j * 16 + lq;
    const float bv = bias ? bias[n] : 0.0f;
#pragma unroll
    for (int r = 0; r < 8; ++r) {
      const int m = tm * 16 + r + half * 8;   // C layout: M in VGPRs, N in lanes
      const float v = (accs[j][r] + bv) * alpha;
      if (mode == 0) {
        o16[(size_t)m * N + n] = (_Float16)v;
      } else if (mode == 1) {
        const int bb = m / vtK, kk = m - bb * vtK;
        o16[((size_t)bb * N + n) * (size_t)vtK + kk] = (_Float16)v;
      } else {
        const size_t idx = (size_t)m * N + n;
        o32[idx] = resid[idx] + v;
      }
    }
  }
}

// ---------------------------------------------------------------------------
// 4) Flash attention, one wave per (b, h, 16-query tile).
//    S^T = K·Q^T per 32-k block (softmax dim lives in registers),
//    ctx^T[64d x 16q] += V^T · P^T via 4 WMMAs. Writes ctx in f16.
// ---------------------------------------------------------------------------
__global__ void __launch_bounds__(128)
attn_kernel(const _Float16* __restrict__ qh, const _Float16* __restrict__ kh,
            const _Float16* __restrict__ vT, const int* __restrict__ iv,
            _Float16* __restrict__ ctx, int Bsz, int Qn, int Kn, int Dn, int Hn) {
  const int lane = threadIdx.x & 31;
  const int wave = threadIdx.x >> 5;
  const int nqt = Qn >> 4;
  const int t = blockIdx.x * (blockDim.x >> 5) + wave;
  if (t >= Bsz * Hn * nqt) return;  // uniform per wave
  const int b = t / (Hn * nqt);
  const int h = (t / nqt) % Hn;
  const int q0 = (t % nqt) << 4;
  const int lq = lane & 15;
  const int half = lane >> 4;
  const int koff = half << 3;       // A-operand K offset (0 or 8)
  const int qrow = q0 + lq;

  // Q as B-operand for both 32-deep d-chunks (DH=64)
  const _Float16* qbase = qh + ((size_t)b * Qn + qrow) * Dn + h * 64;
  const v16h qb0 = ldA16(qbase + koff);
  const v16h qb1 = ldA16(qbase + 32 + koff);

  const int c0 = iv[qrow * 6 + 0], c1 = iv[qrow * 6 + 1];
  const int p0 = iv[qrow * 6 + 2], p1 = iv[qrow * 6 + 3];
  const int n0 = iv[qrow * 6 + 4], n1 = iv[qrow * 6 + 5];

  float mrow = -3.0e38f, lrow = 0.f;
  v8f acc0 = zero8(), acc1 = zero8(), acc2 = zero8(), acc3 = zero8();

  const _Float16* khb = kh + (size_t)b * Kn * Dn + h * 64;
  const _Float16* vtb = vT + ((size_t)b * Dn + h * 64) * (size_t)Kn;

  for (int kb = 0; kb < Kn; kb += 32) {
    // S^T tiles: s0 covers k=kb..kb+15 (register rows), s1 covers kb+16..kb+31
    v8f s0 = zero8(), s1 = zero8();
    {
      const _Float16* ka = khb + (size_t)(kb + lq) * Dn;
      s0 = wmma_f16(ldA16(ka + koff), qb0, s0);
      s0 = wmma_f16(ldA16(ka + 32 + koff), qb1, s0);
      const _Float16* ka2 = khb + (size_t)(kb + 16 + lq) * Dn;
      s1 = wmma_f16(ldA16(ka2 + koff), qb0, s1);
      s1 = wmma_f16(ldA16(ka2 + 32 + koff), qb1, s1);
    }
    float bm = -3.0e38f;
    float e0[8], e1[8];
#pragma unroll
    for (int r = 0; r < 8; ++r) {
      const int k0 = kb + r + koff;           // row index of s0[r]
      s0[r] += maskv(k0, c0, c1, p0, p1, n0, n1);
      s1[r] += maskv(k0 + 16, c0, c1, p0, p1, n0, n1);
      bm = fmaxf(bm, fmaxf(s0[r], s1[r]));
    }
    bm = fmaxf(bm, __shfl_xor(bm, 16));       // combine the two k-half lanes (same q)
    const float mnew = fmaxf(mrow, bm);
    const float corr = __expf(mrow - mnew);
    float ps = 0.f;
#pragma unroll
    for (int r = 0; r < 8; ++r) {
      e0[r] = __expf(s0[r] - mnew);
      e1[r] = __expf(s1[r] - mnew);
      ps += e0[r] + e1[r];
    }
    ps += __shfl_xor(ps, 16);
    lrow = lrow * corr + ps;
    mrow = mnew;
    // Pack P^T into PV B-operand (lanes 0-15: K=kb+{0..7,16..23}; lanes 16-31: kb+{8..15,24..31})
    v16h pb;
#pragma unroll
    for (int r = 0; r < 8; ++r) {
      pb[r] = (_Float16)e0[r];
      pb[r + 8] = (_Float16)e1[r];
    }
#pragma unroll
    for (int r = 0; r < 8; ++r) {
      acc0[r] *= corr; acc1[r] *= corr; acc2[r] *= corr; acc3[r] *= corr;
    }
    // ctx^T[dtile 16 x 16q] += V^T rows (contiguous k) x P^T
    acc0 = wmma_f16(ldA16(vtb + (size_t)(0  + lq) * Kn + kb + koff), pb, acc0);
    acc1 = wmma_f16(ldA16(vtb + (size_t)(16 + lq) * Kn + kb + koff), pb, acc1);
    acc2 = wmma_f16(ldA16(vtb + (size_t)(32 + lq) * Kn + kb + koff), pb, acc2);
    acc3 = wmma_f16(ldA16(vtb + (size_t)(48 + lq) * Kn + kb + koff), pb, acc3);
  }
  const float inv = 1.0f / lrow;
  // C layout: q = lane&15, d = dtile*16 + r + 8*half -> scatter to ctx[b][q][h*64+d]
  _Float16* cbs = ctx + ((size_t)b * Qn + q0 + lq) * Dn + h * 64;
#pragma unroll
  for (int r = 0; r < 8; ++r) {
    const int dr = r + half * 8;
    cbs[0  + dr] = (_Float16)(acc0[r] * inv);
    cbs[16 + dr] = (_Float16)(acc1[r] * inv);
    cbs[32 + dr] = (_Float16)(acc2[r] * inv);
    cbs[48 + dr] = (_Float16)(acc3[r] * inv);
  }
}

// ---------------------------------------------------------------------------
extern "C" void kernel_launch(void* const* d_in, const int* in_sizes, int n_in,
                              void* d_out, int out_size, void* d_ws, size_t ws_size,
                              hipStream_t stream) {
  constexpr int Bc = 2, Qc = 1024, Kc = 4096, Dc = 1024, Hc = 16;
  const float* queries = (const float*)d_in[0];
  const float* keys    = (const float*)d_in[1];
  const float* values  = (const float*)d_in[2];
  const int*   wb      = (const int*)d_in[3];
  const int*   cb      = (const int*)d_in[4];
  const float* gamma   = (const float*)d_in[5];
  const float* beta    = (const float*)d_in[6];
  const float* in_w    = (const float*)d_in[7];
  const float* in_b    = (const float*)d_in[8];
  const float* out_w   = (const float*)d_in[9];
  const float* out_b   = (const float*)d_in[10];
  float* out = (float*)d_out;

  size_t off = 0;
  auto carve = [&](size_t bytes) {
    void* p = (char*)d_ws + off;
    off += (bytes + 255) & ~(size_t)255;
    return p;
  };
  _Float16* x16    = (_Float16*)carve((size_t)Bc * Qc * Dc * 2);
  _Float16* keys16 = (_Float16*)carve((size_t)Bc * Kc * Dc * 2);
  _Float16* vals16 = (_Float16*)carve((size_t)Bc * Kc * Dc * 2);
  _Float16* wqkv16 = (_Float16*)carve((size_t)3 * Dc * Dc * 2);
  _Float16* ow16   = (_Float16*)carve((size_t)Dc * Dc * 2);
  _Float16* qh16   = (_Float16*)carve((size_t)Bc * Qc * Dc * 2);
  _Float16* kh16   = (_Float16*)carve((size_t)Bc * Kc * Dc * 2);
  _Float16* vT16   = (_Float16*)carve((size_t)Bc * Kc * Dc * 2);
  _Float16* ctx16  = (_Float16*)carve((size_t)Bc * Qc * Dc * 2);
  int*      iv     = (int*)     carve((size_t)Qc * 6 * 4);
  (void)ws_size; (void)in_sizes; (void)n_in; (void)out_size;

  auto cvt = [&](const float* src, _Float16* dst, size_t n) {
    cvt_kernel<<<(unsigned)((n / 8 + 255) / 256), 256, 0, stream>>>(src, dst, n);
  };

  // 0) one-time f32 -> f16 conversions
  cvt(keys,   keys16, (size_t)Bc * Kc * Dc);
  cvt(values, vals16, (size_t)Bc * Kc * Dc);
  cvt(in_w,   wqkv16, (size_t)3 * Dc * Dc);
  cvt(out_w,  ow16,   (size_t)Dc * Dc);

  // 1) LayerNorm(queries) -> x16 (f16)
  ln_kernel<<<Bc * Qc, 256, 0, stream>>>(queries, gamma, beta, x16, Dc);

  // 2) mask intervals (independent of b)
  intervals_kernel<<<(Qc + 255) / 256, 256, 0, stream>>>(wb, cb, iv, Qc, Qc / 8);

  const float scale = 0.125f;  // 1/sqrt(DH=64)

  // 3a) Q projection (scale folded): qh16 = (x @ wq^T + bq) * scale
  {
    int M = Bc * Qc, waves = (M / 16) * (Dc / 64);
    gemm_f16_wmma<<<(waves + 3) / 4, 128, 0, stream>>>(
        x16, wqkv16, in_b, scale, M, Dc, Dc, qh16, nullptr, nullptr, 0, 0);
  }
  // 3b) K projection: kh16 = keys @ wk^T + bk
  {
    int M = Bc * Kc, waves = (M / 16) * (Dc / 64);
    gemm_f16_wmma<<<(waves + 3) / 4, 128, 0, stream>>>(
        keys16, wqkv16 + (size_t)Dc * Dc, in_b + Dc, 1.0f, M, Dc, Dc, kh16, nullptr, nullptr, 0, 0);
  }
  // 3c) V projection, stored transposed per (b, h*64+dh): vT16[(b*Dc+n)*Kc + k]
  {
    int M = Bc * Kc, waves = (M / 16) * (Dc / 64);
    gemm_f16_wmma<<<(waves + 3) / 4, 128, 0, stream>>>(
        vals16, wqkv16 + 2 * (size_t)Dc * Dc, in_b + 2 * Dc, 1.0f, M, Dc, Dc, vT16, nullptr, nullptr, 1, Kc);
  }

  // 4) flash attention -> ctx16 (f16)
  {
    int waves = Bc * Hc * (Qc / 16);
    attn_kernel<<<(waves + 3) / 4, 128, 0, stream>>>(qh16, kh16, vT16, iv, ctx16, Bc, Qc, Kc, Dc, Hc);
  }

  // 5) out projection + bias + residual -> d_out (f32)
  {
    int M = Bc * Qc, waves = (M / 16) * (Dc / 64);
    gemm_f16_wmma<<<(waves + 3) / 4, 128, 0, stream>>>(
        ctx16, ow16, out_b, 1.0f, M, Dc, Dc, nullptr, out, queries, 2, 0);
  }
}
